// SelfAttention_12086037971058
// MI455X (gfx1250) — matrix operand order repaired and checked by reading, not compiled, not run on hardware
//
#include <hip/hip_runtime.h>
#include <hip/hip_bf16.h>

#define BATCH 4
#define SEQ   2048
#define DIM   1024
#define HEADS 16
#define HDIM  64
#define BS    (BATCH * SEQ)   // 8192 rows

typedef __attribute__((ext_vector_type(16))) __bf16 v16bf;
typedef __attribute__((ext_vector_type(8)))  __bf16 v8bf;
typedef __attribute__((ext_vector_type(4)))  __bf16 v4bf;
typedef __attribute__((ext_vector_type(8)))  float  v8f;

// ---------- helpers ----------

__device__ __forceinline__ __bf16 f2bf(float f) {
    unsigned u = __builtin_bit_cast(unsigned, f);
    u += 0x7fffu + ((u >> 16) & 1u);          // round-to-nearest-even
    unsigned short h = (unsigned short)(u >> 16);
    return __builtin_bit_cast(__bf16, h);
}

// A fragment (16x32, MxK): lane = row M (lane&15); two 16-byte chunks at
// K offsets {0,16} + 8*(lane>=16).  (ISA 7.12.2, 16-bit A 16x32)
__device__ __forceinline__ v16bf load_a_frag(const __bf16* t, int ld) {
    const int lane = threadIdx.x & 31;
    const __bf16* p = t + (lane & 15) * ld + ((lane >> 4) << 3);
    v8bf lo = *(const v8bf*)(p);
    v8bf hi = *(const v8bf*)(p + 16);
    return __builtin_shufflevector(lo, hi, 0,1,2,3,4,5,6,7,8,9,10,11,12,13,14,15);
}

// B fragment (32x16, KxN) loaded from N-major storage (row n holds K values
// contiguously): lane = col N (lane&15); two 16-byte chunks at K offsets
// {0,16} + 8*(lane>=16) (same k order as A).
__device__ __forceinline__ v16bf load_b_frag(const __bf16* t, int ld) {
    const int lane = threadIdx.x & 31;
    const __bf16* p = t + (lane & 15) * ld + ((lane >> 4) << 3);
    v8bf lo = *(const v8bf*)(p);
    v8bf hi = *(const v8bf*)(p + 16);
    return __builtin_shufflevector(lo, hi, 0,1,2,3,4,5,6,7,8,9,10,11,12,13,14,15);
}

// CDNA5 async DMA: copy 16 bytes global -> LDS, tracked by ASYNCcnt.
// lds_off = low 32 bits of a shared-aperture generic pointer (== wave-relative
// LDS byte address; flat addressing takes the LDS address from the low 32 bits).
__device__ __forceinline__ void async_copy16(void* lds_ptr, const void* gptr) {
    unsigned lds_off = (unsigned)(uintptr_t)lds_ptr;
    unsigned long long ga = (unsigned long long)(uintptr_t)gptr;
    asm volatile("global_load_async_to_lds_b128 %0, %1, off"
                 :: "v"(lds_off), "v"(ga) : "memory");
}
__device__ __forceinline__ void wait_async0() {
    asm volatile("s_wait_asynccnt 0x0" ::: "memory");
}

// ---------- kernel 1: f32 -> bf16 ----------

__global__ __launch_bounds__(256)
void cvt_f32_bf16(const float* __restrict__ in, __bf16* __restrict__ out, int n) {
    int i = (blockIdx.x * 256 + threadIdx.x) * 4;
    if (i + 3 < n) {
        float4 f = *(const float4*)(in + i);
        v4bf o = { f2bf(f.x), f2bf(f.y), f2bf(f.z), f2bf(f.w) };
        *(v4bf*)(out + i) = o;
    }
}

// ---------- kernel 2: QKV projection  out = x @ W^T  (bf16 WMMA) ----------
// grid = (BS/64, DIM/64, 3); block = 128 (4 waves). Each workgroup: 64x64 tile.
// which: 0->Q [b,h,s,d], 1->K [b,h,s,d], 2->V transposed [b,h,d,s].

__global__ __launch_bounds__(128)
void qkv_gemm(const __bf16* __restrict__ xw,
              const __bf16* __restrict__ wqb, const __bf16* __restrict__ wkb,
              const __bf16* __restrict__ wvb,
              __bf16* __restrict__ qb, __bf16* __restrict__ kb,
              __bf16* __restrict__ vt) {
    __shared__ __bf16 As[64][40];   // 64 rows x 32 k, stride 40 (80B, 16B-aligned)
    __shared__ __bf16 Bs[64][40];

    const int tid  = threadIdx.x;
    const int wave = tid >> 5;
    const int lane = tid & 31;
    const int m0 = blockIdx.x * 64;
    const int n0 = blockIdx.y * 64;
    const int which = blockIdx.z;
    const __bf16* W = (which == 0) ? wqb : (which == 1) ? wkb : wvb;

    v8f acc[4];
    v8f zero = {0.f, 0.f, 0.f, 0.f, 0.f, 0.f, 0.f, 0.f};
#pragma unroll
    for (int j = 0; j < 4; ++j) acc[j] = zero;

    const int r = tid >> 1;            // 0..63
    const int c = (tid & 1) * 16;      // 0 or 16

    for (int k0 = 0; k0 < DIM; k0 += 32) {
        // cooperative stage of A (x rows) and B (W rows) tiles
        const __bf16* ga = xw + (size_t)(m0 + r) * DIM + k0 + c;
        const __bf16* gb = W  + (size_t)(n0 + r) * DIM + k0 + c;
        *(v8bf*)&As[r][c]     = *(const v8bf*)(ga);
        *(v8bf*)&As[r][c + 8] = *(const v8bf*)(ga + 8);
        *(v8bf*)&Bs[r][c]     = *(const v8bf*)(gb);
        *(v8bf*)&Bs[r][c + 8] = *(const v8bf*)(gb + 8);
        __syncthreads();

        // preload all fragments, then the WMMA chain (lets ds_loads overlap)
        v16bf a  = load_a_frag(&As[wave * 16][0], 40);
        v16bf b0 = load_b_frag(&Bs[0][0],  40);
        v16bf b1 = load_b_frag(&Bs[16][0], 40);
        v16bf b2 = load_b_frag(&Bs[32][0], 40);
        v16bf b3 = load_b_frag(&Bs[48][0], 40);
        acc[0] = __builtin_amdgcn_wmma_f32_16x16x32_bf16(
            false, a, false, b0, (short)0, acc[0], false, false);
        acc[1] = __builtin_amdgcn_wmma_f32_16x16x32_bf16(
            false, a, false, b1, (short)0, acc[1], false, false);
        acc[2] = __builtin_amdgcn_wmma_f32_16x16x32_bf16(
            false, a, false, b2, (short)0, acc[2], false, false);
        acc[3] = __builtin_amdgcn_wmma_f32_16x16x32_bf16(
            false, a, false, b3, (short)0, acc[3], false, false);
        __syncthreads();
    }

    // scatter C fragments: lane holds col n=(lane&15), rows rr + 8*(lane>=16)
    const int nl   = lane & 15;
    const int half = lane >> 4;
#pragma unroll
    for (int j = 0; j < 4; ++j) {
#pragma unroll
        for (int rr = 0; rr < 8; ++rr) {
            int mg = m0 + wave * 16 + rr + half * 8;     // global row (b,s)
            int ng = n0 + j * 16 + nl;                   // global col (h,d)
            int b = mg >> 11, s = mg & (SEQ - 1);
            int h = ng >> 6,  d = ng & (HDIM - 1);
            __bf16 v = f2bf(acc[j][rr]);
            if (which == 0)
                qb[(((size_t)b * HEADS + h) * SEQ + s) * HDIM + d] = v;
            else if (which == 1)
                kb[(((size_t)b * HEADS + h) * SEQ + s) * HDIM + d] = v;
            else
                vt[(((size_t)b * HEADS + h) * HDIM + d) * SEQ + s] = v;
        }
    }
}

// ---------- kernel 3: flash attention ----------
// grid = (SEQ/64, B*H); block = 128 (4 waves). Wave w owns 16 query rows,
// full 16x64 f32 accumulator; loop over 32-key tiles with online softmax.
// K/V tiles staged with GLOBAL_LOAD_ASYNC_TO_LDS_B128 (ASYNCcnt).

__global__ __launch_bounds__(128)
void flash_attn(const __bf16* __restrict__ qb, const __bf16* __restrict__ kb,
                const __bf16* __restrict__ vt, float* __restrict__ out) {
    __shared__ __bf16 Ks[32][72];       // 32 keys x 64 d (stride 72 -> 144B)
    __shared__ __bf16 Vs[64][40];       // 64 d   x 32 s (stride 40 -> 80B)
    __shared__ __bf16 Ps[4][16][40];    // per-wave P tile 16x32

    const int tid  = threadIdx.x;
    const int wave = tid >> 5;
    const int lane = tid & 31;
    const int bh = blockIdx.y;
    const int b = bh >> 4, h = bh & 15;
    const int q0 = blockIdx.x * 64;

    const __bf16* Q = qb + ((size_t)bh * SEQ + q0 + wave * 16) * HDIM;
    const __bf16* K = kb + (size_t)bh * SEQ * HDIM;
    const __bf16* V = vt + (size_t)bh * HDIM * SEQ;

    // Q fragments held in registers for whole pass (d 0..31 and 32..63)
    v16bf qa0 = load_a_frag(Q, HDIM);
    v16bf qa1 = load_a_frag(Q + 32, HDIM);

    v8f acc[4];
    v8f zero = {0.f, 0.f, 0.f, 0.f, 0.f, 0.f, 0.f, 0.f};
#pragma unroll
    for (int j = 0; j < 4; ++j) acc[j] = zero;
    float mx[8], lsum[8];
#pragma unroll
    for (int rr = 0; rr < 8; ++rr) { mx[rr] = -3.0e38f; lsum[rr] = 0.f; }

    const float scale = 0.03125f;   // 1/sqrt(1024)
    const int kr = tid >> 2, kc = (tid & 3) * 16;   // K tile staging coords
    const int vr = tid >> 1, vc = (tid & 1) * 16;   // V tile staging coords
    const int nl = lane & 15, half = lane >> 4;

    for (int j0 = 0; j0 < SEQ; j0 += 32) {
        // async DMA stage: K (32 keys x 64d) and V^T (64d x 32s) tiles
        const __bf16* gk = K + (size_t)(j0 + kr) * HDIM + kc;
        const __bf16* gv = V + (size_t)vr * SEQ + j0 + vc;
        async_copy16(&Ks[kr][kc],     gk);
        async_copy16(&Ks[kr][kc + 8], gk + 8);
        async_copy16(&Vs[vr][vc],     gv);
        async_copy16(&Vs[vr][vc + 8], gv + 8);
        if (j0 + 32 < SEQ) {     // global_prefetch_b8 for next tiles
            __builtin_prefetch(K + (size_t)(j0 + 32 + kr) * HDIM + kc, 0, 1);
            __builtin_prefetch(V + (size_t)vr * SEQ + j0 + 32 + vc, 0, 1);
        }
        wait_async0();
        __syncthreads();

        // S = (Q K^T) * scale : two 16x16 C frags (keys j0..+15, j0+16..+31)
        v16bf kb00 = load_b_frag(&Ks[0][0],   72);
        v16bf kb01 = load_b_frag(&Ks[0][32],  72);
        v16bf kb10 = load_b_frag(&Ks[16][0],  72);
        v16bf kb11 = load_b_frag(&Ks[16][32], 72);
        v8f s0 = zero, s1 = zero;
        s0 = __builtin_amdgcn_wmma_f32_16x16x32_bf16(
            false, qa0, false, kb00, (short)0, s0, false, false);
        s0 = __builtin_amdgcn_wmma_f32_16x16x32_bf16(
            false, qa1, false, kb01, (short)0, s0, false, false);
        s1 = __builtin_amdgcn_wmma_f32_16x16x32_bf16(
            false, qa0, false, kb10, (short)0, s1, false, false);
        s1 = __builtin_amdgcn_wmma_f32_16x16x32_bf16(
            false, qa1, false, kb11, (short)0, s1, false, false);

        // online softmax: rows live across 16-lane halves of the wave
#pragma unroll
        for (int rr = 0; rr < 8; ++rr) {
            float a0 = s0[rr] * scale, a1 = s1[rr] * scale;
            float t = fmaxf(a0, a1);
            t = fmaxf(t, __shfl_xor(t, 1));
            t = fmaxf(t, __shfl_xor(t, 2));
            t = fmaxf(t, __shfl_xor(t, 4));
            t = fmaxf(t, __shfl_xor(t, 8));
            float mn = fmaxf(mx[rr], t);
            float corr = __expf(mx[rr] - mn);
            mx[rr] = mn;
            float e0 = __expf(a0 - mn), e1 = __expf(a1 - mn);
            float rs = e0 + e1;
            rs += __shfl_xor(rs, 1);
            rs += __shfl_xor(rs, 2);
            rs += __shfl_xor(rs, 4);
            rs += __shfl_xor(rs, 8);
            lsum[rr] = lsum[rr] * corr + rs;
#pragma unroll
            for (int j = 0; j < 4; ++j) acc[j][rr] *= corr;
            int m = rr + half * 8;
            Ps[wave][m][nl]      = f2bf(e0);   // C-layout -> LDS 16x32 tile
            Ps[wave][m][nl + 16] = f2bf(e1);
        }
        // same-wave LDS RAW: wait stores, then reload P in A-fragment layout
        asm volatile("s_wait_dscnt 0" ::: "memory");
        v16bf pa = load_a_frag(&Ps[wave][0][0], 40);
        v16bf vb0 = load_b_frag(&Vs[0][0],  40);
        v16bf vb1 = load_b_frag(&Vs[16][0], 40);
        v16bf vb2 = load_b_frag(&Vs[32][0], 40);
        v16bf vb3 = load_b_frag(&Vs[48][0], 40);
        acc[0] = __builtin_amdgcn_wmma_f32_16x16x32_bf16(
            false, pa, false, vb0, (short)0, acc[0], false, false);
        acc[1] = __builtin_amdgcn_wmma_f32_16x16x32_bf16(
            false, pa, false, vb1, (short)0, acc[1], false, false);
        acc[2] = __builtin_amdgcn_wmma_f32_16x16x32_bf16(
            false, pa, false, vb2, (short)0, acc[2], false, false);
        acc[3] = __builtin_amdgcn_wmma_f32_16x16x32_bf16(
            false, pa, false, vb3, (short)0, acc[3], false, false);
        __syncthreads();
    }

    // epilogue: out = acc / l
    float inv[8];
#pragma unroll
    for (int rr = 0; rr < 8; ++rr) inv[rr] = 1.0f / lsum[rr];
#pragma unroll
    for (int dt = 0; dt < 4; ++dt) {
#pragma unroll
        for (int rr = 0; rr < 8; ++rr) {
            int s = q0 + wave * 16 + rr + half * 8;
            int d = dt * 16 + nl;
            out[((size_t)b * SEQ + s) * DIM + h * HDIM + d] = acc[dt][rr] * inv[rr];
        }
    }
}

// ---------- launch ----------

extern "C" void kernel_launch(void* const* d_in, const int* in_sizes, int n_in,
                              void* d_out, int out_size, void* d_ws, size_t ws_size,
                              hipStream_t stream) {
    (void)in_sizes; (void)n_in; (void)out_size;
    const float* x  = (const float*)d_in[0];
    const float* Wq = (const float*)d_in[1];
    const float* Wk = (const float*)d_in[2];
    const float* Wv = (const float*)d_in[3];
    float* out = (float*)d_out;

    char* ws = (char*)d_ws;
    size_t off = 0;
    __bf16* xw  = (__bf16*)(ws + off); off += (size_t)BS * DIM * 2;
    __bf16* wqb = (__bf16*)(ws + off); off += (size_t)DIM * DIM * 2;
    __bf16* wkb = (__bf16*)(ws + off); off += (size_t)DIM * DIM * 2;
    __bf16* wvb = (__bf16*)(ws + off); off += (size_t)DIM * DIM * 2;
    __bf16* qb  = (__bf16*)(ws + off); off += (size_t)BS * DIM * 2;
    __bf16* kb  = (__bf16*)(ws + off); off += (size_t)BS * DIM * 2;
    __bf16* vt  = (__bf16*)(ws + off); off += (size_t)BS * DIM * 2;
    if (ws_size < off) return;

    cvt_f32_bf16<<<(BS * DIM) / 1024, 256, 0, stream>>>(x, xw, BS * DIM);
    cvt_f32_bf16<<<(DIM * DIM) / 1024, 256, 0, stream>>>(Wq, wqb, DIM * DIM);
    cvt_f32_bf16<<<(DIM * DIM) / 1024, 256, 0, stream>>>(Wk, wkb, DIM * DIM);
    cvt_f32_bf16<<<(DIM * DIM) / 1024, 256, 0, stream>>>(Wv, wvb, DIM * DIM);

    qkv_gemm<<<dim3(BS / 64, DIM / 64, 3), 128, 0, stream>>>(
        xw, wqb, wkb, wvb, qb, kb, vt);

    flash_attn<<<dim3(SEQ / 64, BATCH * HEADS), 128, 0, stream>>>(
        qb, kb, vt, out);
}
